// SAE_D_62010737819897
// MI455X (gfx1250) — matrix operand order repaired and checked by reading, not compiled, not run on hardware
//
#include <hip/hip_runtime.h>

typedef __attribute__((ext_vector_type(16))) __bf16 v16bf;
typedef __attribute__((ext_vector_type(8)))  __bf16 v8bf;
typedef __attribute__((ext_vector_type(8)))  float  v8f;

constexpr int NB = 4096;   // batch
constexpr int ND = 1024;   // embedding dim
constexpr int NH = 16384;  // hidden dim
constexpr int NK = 32;     // top-k

// encoder tiling
constexpr int BT_M = 64;   // block tile M
constexpr int BT_N = 128;  // block tile N
constexpr int KT   = 32;   // K chunk (one bf16 WMMA depth)
constexpr int LDA  = 40;   // padded bf16 row stride (80B rows -> 16B aligned frags)
constexpr int LDB  = 40;

union ABfrag { v16bf v; v8bf h[2]; };

__device__ __forceinline__ void bf16_split(float x, __bf16& hi, __bf16& lo) {
  hi = (__bf16)x;
  lo = (__bf16)(x - (float)hi);
}

// C[B,H] = relu(X[B,D] @ W[D,H] + bias) for both streams (blockIdx.z selects).
// fp32 emulated via bf16 hi/lo split: acc += Ah*Bh + Ah*Bl + Al*Bh  (3 WMMAs / K=32)
__global__ __launch_bounds__(256) void sae_encoder_bf16x3(
    const float* __restrict__ Xv, const float* __restrict__ Xt,
    const float* __restrict__ Wv, const float* __restrict__ Wt,
    const float* __restrict__ bv, const float* __restrict__ bt,
    float* __restrict__ latent)
{
  const int z = blockIdx.z;
  const float* __restrict__ X    = z ? Xt : Xv;
  const float* __restrict__ W    = z ? Wt : Wv;
  const float* __restrict__ bias = z ? bt : bv;
  float* __restrict__ out = latent + (size_t)z * NB * NH;

  __shared__ __bf16 Ah[BT_M * LDA];
  __shared__ __bf16 Al[BT_M * LDA];
  __shared__ __bf16 Bh[BT_N * LDB];   // stored transposed: [n][k]
  __shared__ __bf16 Bl[BT_N * LDB];

  const int tid  = threadIdx.x;
  const int lane = tid & 31;
  const int wave = tid >> 5;      // 8 waves
  const int wm   = wave >> 2;     // 0..1 -> M offset 32*wm
  const int wn   = wave & 3;      // 0..3 -> N offset 32*wn

  const int m0 = blockIdx.y * BT_M;
  const int n0 = blockIdx.x * BT_N;

  v8f acc[2][2];
#pragma unroll
  for (int mi = 0; mi < 2; ++mi)
#pragma unroll
    for (int ni = 0; ni < 2; ++ni)
#pragma unroll
      for (int e = 0; e < 8; ++e) acc[mi][ni][e] = 0.0f;

  const int l16   = lane & 15;   // A: M row / B: N col / C: N col
  const int khalf = lane >> 4;   // K half select per ISA layout

  for (int d0 = 0; d0 < ND; d0 += KT) {
    // ---- stage A tile (64 x 32 f32) -> LDS bf16 hi/lo, [m][k]
#pragma unroll
    for (int i = 0; i < 2; ++i) {
      const int lin = tid + 256 * i;        // 0..511
      const int m   = lin >> 3;
      const int kv  = (lin & 7) << 2;
      const float4 x4 = *(const float4*)(X + (size_t)(m0 + m) * ND + d0 + kv);
      __bf16 h0,l0,h1,l1,h2,l2,h3,l3;
      bf16_split(x4.x, h0, l0); bf16_split(x4.y, h1, l1);
      bf16_split(x4.z, h2, l2); bf16_split(x4.w, h3, l3);
      __bf16* ah = &Ah[m * LDA + kv];
      __bf16* al = &Al[m * LDA + kv];
      ah[0]=h0; ah[1]=h1; ah[2]=h2; ah[3]=h3;
      al[0]=l0; al[1]=l1; al[2]=l2; al[3]=l3;
    }
    // ---- stage B tile (32 x 128 f32) -> LDS bf16 hi/lo transposed [n][k]
#pragma unroll
    for (int i = 0; i < 4; ++i) {
      const int lin = tid + 256 * i;        // 0..1023
      const int k   = lin >> 5;
      const int nv  = (lin & 31) << 2;
      const float4 w4 = *(const float4*)(W + (size_t)(d0 + k) * NH + n0 + nv);
      __bf16 h0,l0,h1,l1,h2,l2,h3,l3;
      bf16_split(w4.x, h0, l0); bf16_split(w4.y, h1, l1);
      bf16_split(w4.z, h2, l2); bf16_split(w4.w, h3, l3);
      Bh[(nv + 0) * LDB + k] = h0; Bl[(nv + 0) * LDB + k] = l0;
      Bh[(nv + 1) * LDB + k] = h1; Bl[(nv + 1) * LDB + k] = l1;
      Bh[(nv + 2) * LDB + k] = h2; Bl[(nv + 2) * LDB + k] = l2;
      Bh[(nv + 3) * LDB + k] = h3; Bl[(nv + 3) * LDB + k] = l3;
    }
    if (d0 + KT < ND) {  // warm next K chunk (global_prefetch_b8)
      __builtin_prefetch(X + (size_t)(m0 + (tid >> 3)) * ND + d0 + KT, 0, 1);
      __builtin_prefetch(W + (size_t)(d0 + KT + (tid >> 5)) * NH + n0 + ((tid & 31) << 2), 0, 1);
    }
    __syncthreads();

    // ---- fragments (ISA 16-bit A 16x32 / B 32x16 layouts)
    ABfrag ahf[2], alf[2], bhf[2], blf[2];
#pragma unroll
    for (int mi = 0; mi < 2; ++mi) {
      const int base = (wm * 32 + mi * 16 + l16) * LDA + khalf * 8;
      ahf[mi].h[0] = *(const v8bf*)&Ah[base];
      ahf[mi].h[1] = *(const v8bf*)&Ah[base + 16];
      alf[mi].h[0] = *(const v8bf*)&Al[base];
      alf[mi].h[1] = *(const v8bf*)&Al[base + 16];
    }
#pragma unroll
    for (int ni = 0; ni < 2; ++ni) {
      const int base = (wn * 32 + ni * 16 + l16) * LDB + khalf * 8;
      bhf[ni].h[0] = *(const v8bf*)&Bh[base];
      bhf[ni].h[1] = *(const v8bf*)&Bh[base + 16];
      blf[ni].h[0] = *(const v8bf*)&Bl[base];
      blf[ni].h[1] = *(const v8bf*)&Bl[base + 16];
    }

#pragma unroll
    for (int mi = 0; mi < 2; ++mi)
#pragma unroll
      for (int ni = 0; ni < 2; ++ni) {
        acc[mi][ni] = __builtin_amdgcn_wmma_f32_16x16x32_bf16(
            false, ahf[mi].v, false, bhf[ni].v, (short)0, acc[mi][ni], false, false);
        acc[mi][ni] = __builtin_amdgcn_wmma_f32_16x16x32_bf16(
            false, ahf[mi].v, false, blf[ni].v, (short)0, acc[mi][ni], false, false);
        acc[mi][ni] = __builtin_amdgcn_wmma_f32_16x16x32_bf16(
            false, alf[mi].v, false, bhf[ni].v, (short)0, acc[mi][ni], false, false);
      }
    __syncthreads();
  }

  // ---- epilogue: bias + ReLU, dense store into latent region
#pragma unroll
  for (int ni = 0; ni < 2; ++ni) {
    const int ngl = n0 + wn * 32 + ni * 16 + l16;
    const float bb = bias[ngl];
#pragma unroll
    for (int mi = 0; mi < 2; ++mi) {
#pragma unroll
      for (int e = 0; e < 8; ++e) {
        const int mgl = m0 + wm * 32 + mi * 16 + khalf * 8 + e;
        const float vv = acc[mi][ni][e] + bb;
        out[(size_t)mgl * NH + ngl] = vv > 0.0f ? vv : 0.0f;
      }
    }
  }
}

// Exact top-K per row via 4-round radix select on float bits (all values >= 0),
// row held in registers; sparsifies latent in place and compacts K pairs to ws.
__global__ __launch_bounds__(256) void sae_topk(
    float* __restrict__ latent, int* __restrict__ idx_ws, float* __restrict__ val_ws)
{
  const int z   = blockIdx.y;
  const int row = blockIdx.x;
  float* __restrict__ hrow = latent + ((size_t)z * NB + row) * NH;
  const size_t pbase = ((size_t)z * NB + row) * NK;
  const int tid = threadIdx.x;

  float v[NH / 256];
#pragma unroll
  for (int i = 0; i < NH / 256; ++i) v[i] = hrow[i * 256 + tid];

  __shared__ unsigned hist[256];
  __shared__ unsigned s_bin, s_need, tieTick, outPos;

  unsigned prefix = 0;
  unsigned need = NK;

  for (int shift = 24; shift >= 0; shift -= 8) {
    hist[tid] = 0;
    __syncthreads();
    const unsigned hmask = (shift == 24) ? 0u : (~0u << (shift + 8));
#pragma unroll
    for (int i = 0; i < NH / 256; ++i) {
      const unsigned u = __float_as_uint(v[i]);
      if ((u & hmask) == prefix) atomicAdd(&hist[(u >> shift) & 0xFF], 1u);
    }
    __syncthreads();
    if (tid == 0) {
      unsigned cum = 0; int bsel = 0;
      for (int b2 = 255; b2 >= 0; --b2) {
        const unsigned c = hist[b2];
        if (cum + c >= need) { bsel = b2; break; }
        cum += c;
      }
      s_bin  = (unsigned)bsel;
      s_need = need - cum;
    }
    __syncthreads();
    prefix |= s_bin << shift;
    need = s_need;
    __syncthreads();
  }

  const unsigned thr  = prefix;   // value of K-th largest
  const unsigned ties = need;     // # of ==thr elements to keep
  if (tid == 0) { tieTick = 0; outPos = 0; }
  __syncthreads();

#pragma unroll
  for (int i = 0; i < NH / 256; ++i) {
    const int e = i * 256 + tid;
    const unsigned u = __float_as_uint(v[i]);
    bool keep = (u > thr);
    if (!keep && u == thr) keep = (atomicAdd(&tieTick, 1u) < ties);
    hrow[e] = keep ? v[i] : 0.0f;
    if (keep) {
      const unsigned p = atomicAdd(&outPos, 1u);
      idx_ws[pbase + p] = e;
      val_ws[pbase + p] = v[i];
    }
  }
}

// recon[b,:] = bias + sum_j val_j * Wdec[idx_j,:]  (K=32 gathers, L2-resident Wdec)
__global__ __launch_bounds__(256) void sae_decoder(
    const float* __restrict__ Wvd, const float* __restrict__ Wtd,
    const float* __restrict__ bvd, const float* __restrict__ btd,
    const int* __restrict__ idx_ws, const float* __restrict__ val_ws,
    float* __restrict__ recon)
{
  const int z   = blockIdx.y;
  const int row = blockIdx.x;
  const float* __restrict__ Wd = z ? Wtd : Wvd;
  const float* __restrict__ bd = z ? btd : bvd;
  float* __restrict__ out = recon + ((size_t)z * NB + row) * ND;
  const size_t pbase = ((size_t)z * NB + row) * NK;

  __shared__ int   sidx[NK];
  __shared__ float sval[NK];
  const int tid = threadIdx.x;
  if (tid < NK) { sidx[tid] = idx_ws[pbase + tid]; sval[tid] = val_ws[pbase + tid]; }
  __syncthreads();

  const int c = tid << 2;                 // 256 threads x 4 cols = 1024
  float4 a4 = *(const float4*)(bd + c);
#pragma unroll 8
  for (int j = 0; j < NK; ++j) {
    const float4 w4 = *(const float4*)(Wd + (size_t)sidx[j] * ND + c);
    const float s = sval[j];
    a4.x += s * w4.x; a4.y += s * w4.y; a4.z += s * w4.z; a4.w += s * w4.w;
  }
  *(float4*)(out + c) = a4;
}

extern "C" void kernel_launch(void* const* d_in, const int* in_sizes, int n_in,
                              void* d_out, int out_size, void* d_ws, size_t ws_size,
                              hipStream_t stream) {
  (void)in_sizes; (void)n_in; (void)out_size; (void)ws_size;
  const float* Xv     = (const float*)d_in[0];
  const float* Xt     = (const float*)d_in[1];
  const float* Wv_enc = (const float*)d_in[2];
  const float* bv_enc = (const float*)d_in[3];
  const float* Wt_enc = (const float*)d_in[4];
  const float* bt_enc = (const float*)d_in[5];
  const float* Wv_dec = (const float*)d_in[6];
  const float* bv_dec = (const float*)d_in[7];
  const float* Wt_dec = (const float*)d_in[8];
  const float* bt_dec = (const float*)d_in[9];

  float* out    = (float*)d_out;
  float* recon  = out;                          // [2][B][D]
  float* latent = out + (size_t)2 * NB * ND;    // [2][B][H]

  int*   idx_ws = (int*)d_ws;                                             // 2*B*K ints
  float* val_ws = (float*)((char*)d_ws + (size_t)2 * NB * NK * sizeof(int)); // 2*B*K floats

  dim3 gEnc(NH / BT_N, NB / BT_M, 2);   // (128, 64, 2)
  sae_encoder_bf16x3<<<gEnc, 256, 0, stream>>>(Xv, Xt, Wv_enc, Wt_enc, bv_enc, bt_enc, latent);

  dim3 gRow(NB, 2);
  sae_topk<<<gRow, 256, 0, stream>>>(latent, idx_ws, val_ws);
  sae_decoder<<<gRow, 256, 0, stream>>>(Wv_dec, Wt_dec, bv_dec, bt_dec, idx_ws, val_ws, recon);
}
